// EdgeBlock_62070867362421
// MI455X (gfx1250) — matrix operand order repaired and checked by reading, not compiled, not run on hardware
//
#include <hip/hip_runtime.h>

// EdgeBlock GNN: out = relu([ef | nf[snd] | nf[rcv]] @ W1 + b1) @ W2 + b2
// f16 WMMA (v_wmma_f32_16x16x32_f16) with f32 accumulate; memory-bound design.

typedef _Float16 v16h __attribute__((ext_vector_type(16)));
typedef _Float16 v8h  __attribute__((ext_vector_type(8)));
typedef _Float16 v4h  __attribute__((ext_vector_type(4)));
typedef _Float16 v2h  __attribute__((ext_vector_type(2)));
typedef float    v8f  __attribute__((ext_vector_type(8)));

#define TILE_M   128      // edges per workgroup tile
#define THREADS  256      // 8 waves (wave32)
#define D_IN     96
#define D_HID    128
#define D_OUT    32
#define XH_LD    136      // halves/row for X|H union tile: 272B = 17*16B, bank-friendly
#define W1_LD    104      // halves/row for W1^T: 208B = 13*16B
#define W2_LD    136      // halves/row for W2^T
#define GITER    ((TILE_M * 24) / THREADS)   // 12 gather float4's per thread

static __device__ inline v16h cat8(v8h a, v8h b) {
  return __builtin_shufflevector(a, b, 0,1,2,3,4,5,6,7,8,9,10,11,12,13,14,15);
}

__global__ __launch_bounds__(THREADS)
void edgeblock_kernel(const float* __restrict__ ef, const float* __restrict__ nf,
                      const int* __restrict__ snd, const int* __restrict__ rcv,
                      const float* __restrict__ W1, const float* __restrict__ b1,
                      const float* __restrict__ W2, const float* __restrict__ b2,
                      float* __restrict__ out, int nTiles)
{
  __shared__ _Float16 XH [TILE_M * XH_LD];   // X (96 cols) then overwritten by H (128 cols)
  __shared__ _Float16 W1T[D_HID  * W1_LD];   // W1^T: [n][k], f16
  __shared__ _Float16 W2T[D_OUT  * W2_LD];   // W2^T: [n][k], f16
  __shared__ float    b1s[D_HID];
  __shared__ float    b2s[D_OUT];
  __shared__ int      idx2[2 * TILE_M];      // [0:128) senders, [128:256) receivers

  const int tid  = threadIdx.x;
  const int wave = tid >> 5;
  const int lane = tid & 31;
  const int l16  = lane & 15;
  const int hi   = lane >> 4;          // 0: lanes 0-15, 1: lanes 16-31
  const int rowBase = wave * 16;       // this wave's 16 edge rows within the tile

  // One-time: transpose + convert weights into LDS (amortized over grid-stride tiles)
  for (int i = tid; i < D_IN * D_HID; i += THREADS) {
    int k = i >> 7, n = i & (D_HID - 1);          // W1 is [96][128] row-major
    W1T[n * W1_LD + k] = (_Float16)W1[i];
  }
  for (int i = tid; i < D_HID * D_OUT; i += THREADS) {
    int k = i >> 5, n = i & (D_OUT - 1);          // W2 is [128][32] row-major
    W2T[n * W2_LD + k] = (_Float16)W2[i];
  }
  if (tid < D_HID) b1s[tid] = b1[tid];
  if (tid < D_OUT) b2s[tid] = b2[tid];

  for (int tile = blockIdx.x; tile < nTiles; tile += gridDim.x) {
    const int e0 = tile * TILE_M;
    __syncthreads();   // weights ready (iter 0); XH/idx2 free of prior-iteration readers

    {
      const int* ip = (tid < TILE_M) ? (snd + e0 + tid) : (rcv + e0 + tid - TILE_M);
      idx2[tid] = *ip;
    }
    __syncthreads();

    // Gather: X[e][0:96] = [edge_feats | nf[sender] | nf[receiver]], f32 -> f16.
    // Phase 1: issue all 12 full-EXEC b128 loads (branchless source select) so
    // they are in flight together. Phase 2: convert + store to LDS.
    float4 vals[GITER];
    int    laddr[GITER];
#pragma unroll
    for (int it = 0; it < GITER; ++it) {
      int i = tid + it * THREADS;
      int e = i / 24;
      int c = i - e * 24;                                    // float4 chunk 0..23
      unsigned node = (unsigned)idx2[e + ((c < 16) ? 0 : TILE_M)];
      const float* p = (c < 8)
          ? (ef + ((unsigned)(e0 + e) * 32u + (unsigned)c * 4u))
          : (nf + (node * 32u + (unsigned)(c & 7) * 4u));
      laddr[it] = e * XH_LD + c * 4;
      vals[it] = *(const float4*)p;
    }
#pragma unroll
    for (int it = 0; it < GITER; ++it) {
      float4 v = vals[it];
      v4h h = { (_Float16)v.x, (_Float16)v.y, (_Float16)v.z, (_Float16)v.w };
      *(v4h*)&XH[laddr[it]] = h;
    }
    __syncthreads();

    // ---- GEMM1: H = relu(X @ W1 + b1), M=16 (per wave), K=96, N=128 ----
    // A-fragment (f16 16x32): lane<16 holds row l16, K{32c..+7, 32c+16..+23};
    // lane>=16 holds row l16, K{32c+8..+15, 32c+24..+31}.
    const _Float16* xrow = &XH[(rowBase + l16) * XH_LD];
    v16h A1[3];
#pragma unroll
    for (int c = 0; c < 3; ++c) {
      v8h lo = *(const v8h*)(xrow + 32 * c + 8 * hi);
      v8h hh = *(const v8h*)(xrow + 32 * c + 16 + 8 * hi);
      A1[c] = cat8(lo, hh);
    }

    const v2h zz = {};
    // j-blocks of 4: four independent accumulator chains for WMMA ILP without
    // blowing past 256 VGPRs (round-2 lesson).
#pragma unroll
    for (int jb = 0; jb < 8; jb += 4) {
      v8f acc[4];
#pragma unroll
      for (int u = 0; u < 4; ++u) {
        float b = b1s[(jb + u) * 16 + l16];
        acc[u] = (v8f){ b, b, b, b, b, b, b, b };
      }
#pragma unroll
      for (int c = 0; c < 3; ++c) {
#pragma unroll
        for (int u = 0; u < 4; ++u) {
          const _Float16* wrow = &W1T[((jb + u) * 16 + l16) * W1_LD + 8 * hi];
          v8h lo = *(const v8h*)(wrow + 32 * c);
          v8h hh = *(const v8h*)(wrow + 32 * c + 16);
          v16h B = cat8(lo, hh);
          acc[u] = __builtin_amdgcn_wmma_f32_16x16x32_f16(
                     false, A1[c], false, B, (short)0, acc[u], false, false);
        }
      }
      // Convert to f16 (canonical), then packed ReLU (v_pk_max_num_f16).
      // C layout: VGPR r -> row (r + 8*hi), col l16 (+16j).
#pragma unroll
      for (int u = 0; u < 4; ++u) {
#pragma unroll
        for (int r = 0; r < 8; r += 2) {
          v2h p = { (_Float16)acc[u][r], (_Float16)acc[u][r + 1] };
          p = __builtin_elementwise_max(p, zz);
          _Float16* hb = &XH[(rowBase + r + 8 * hi) * XH_LD + (jb + u) * 16 + l16];
          hb[0]     = p.x;
          hb[XH_LD] = p.y;
        }
      }
    }

    // ---- GEMM2: out = H @ W2 + b2, M=16, K=128, N=32 ----
    v16h A2[4];
#pragma unroll
    for (int c = 0; c < 4; ++c) {
      v8h lo = *(const v8h*)(xrow + 32 * c + 8 * hi);   // xrow now holds H row
      v8h hh = *(const v8h*)(xrow + 32 * c + 16 + 8 * hi);
      A2[c] = cat8(lo, hh);
    }

    {
      v8f acc[2];
#pragma unroll
      for (int j = 0; j < 2; ++j) {
        float b = b2s[j * 16 + l16];
        acc[j] = (v8f){ b, b, b, b, b, b, b, b };
      }
#pragma unroll
      for (int c = 0; c < 4; ++c) {
#pragma unroll
        for (int j = 0; j < 2; ++j) {
          const _Float16* wrow = &W2T[(j * 16 + l16) * W2_LD + 8 * hi];
          v8h lo = *(const v8h*)(wrow + 32 * c);
          v8h hh = *(const v8h*)(wrow + 32 * c + 16);
          v16h B = cat8(lo, hh);
          acc[j] = __builtin_amdgcn_wmma_f32_16x16x32_f16(
                     false, A2[c], false, B, (short)0, acc[j], false, false);
        }
      }
      // f32 stores: lanes 0-15 / 16-31 each cover 64B-contiguous row segments
#pragma unroll
      for (int j = 0; j < 2; ++j) {
        float* op = out + (size_t)(e0 + rowBase + 8 * hi) * D_OUT + j * 16 + l16;
#pragma unroll
        for (int r = 0; r < 8; ++r) op[(size_t)r * D_OUT] = acc[j][r];
      }
    }
  }
}

extern "C" void kernel_launch(void* const* d_in, const int* in_sizes, int n_in,
                              void* d_out, int out_size, void* d_ws, size_t ws_size,
                              hipStream_t stream) {
  const float* ef = (const float*)d_in[0];
  const float* nf = (const float*)d_in[1];
  const int*  snd = (const int*)d_in[2];
  const int*  rcv = (const int*)d_in[3];
  const float* W1 = (const float*)d_in[4];
  const float* b1 = (const float*)d_in[5];
  const float* W2 = (const float*)d_in[6];
  const float* b2 = (const float*)d_in[7];
  float* out = (float*)d_out;

  const int E = in_sizes[2];             // 1,600,000 (divisible by TILE_M)
  const int nTiles = E / TILE_M;         // 12,500
  int grid = nTiles < 1280 ? nTiles : 1280;   // persistent-ish: amortize weight staging
  edgeblock_kernel<<<dim3(grid), dim3(THREADS), 0, stream>>>(
      ef, nf, snd, rcv, W1, b1, W2, b2, out, nTiles);
}
